// TreensformerBlockV7_17239998726538
// MI455X (gfx1250) — matrix operand order repaired and checked by reading, not compiled, not run on hardware
//
#include <hip/hip_runtime.h>
#include <hip/hip_bf16.h>

typedef __attribute__((ext_vector_type(16))) _Float16 v16h;
typedef __attribute__((ext_vector_type(8)))  _Float16 v8h;
typedef __attribute__((ext_vector_type(8)))  float    v8f;

#define L_LEVELS 6
#define HGRID    32
#define RDIM     128
#define NEMBD    768
#define HID      3072
#define NH       8
#define HD       16
#define BATCH    8
#define MNODES   1365
#define MP       1376                 // MNODES padded to multiple of 16
#define ROWS_QKV (BATCH*MP)           // 11008 (multiple of 64)
#define ROWS_MLP (BATCH*HGRID*HGRID)  // 8192  (multiple of 64)
#define MAXNB    16

// ---------------- weight transpose (f32 KxN -> f16 NxK) ----------------
__global__ void k_transpose_f16(const float* __restrict__ W, _Float16* __restrict__ Wt,
                                int K, int N) {
  int idx = blockIdx.x * 256 + threadIdx.x;
  if (idx >= K * N) return;
  int k = idx / N, n = idx % N;
  Wt[(size_t)n * K + k] = (_Float16)W[idx];
}

__global__ void k_concat_bias(const float* a, const float* b, const float* c, float* out) {
  int t = blockIdx.x * 128 + threadIdx.x;
  if (t < 128)      out[t] = a[t];
  else if (t < 256) out[t] = b[t - 128];
  else if (t < 384) out[t] = c[t - 256];
}

// ---------------- LN1 + gather unique nodes -> f16 ----------------
__global__ void k_ln_gather(const float* __restrict__ x, const float* __restrict__ g,
                            const float* __restrict__ bta,
                            const int* __restrict__ rep_h, const int* __restrict__ rep_w,
                            const int* __restrict__ rep_l, _Float16* __restrict__ uniq) {
  int bm = blockIdx.x, b = bm / MP, m = bm % MP, t = threadIdx.x;
  if (m >= MNODES) { uniq[(size_t)bm * RDIM + t] = (_Float16)0.f; return; }
  int h = rep_h[m], w = rep_w[m], l = rep_l[m];
  size_t base = ((((size_t)b * HGRID + h) * HGRID + w) * L_LEVELS + l) * RDIM;
  float v = x[base + t];
  __shared__ float red[RDIM];
  red[t] = v; __syncthreads();
  for (int s = 64; s > 0; s >>= 1) { if (t < s) red[t] += red[t + s]; __syncthreads(); }
  float mean = red[0] / RDIM; __syncthreads();
  float d = v - mean; red[t] = d * d; __syncthreads();
  for (int s = 64; s > 0; s >>= 1) { if (t < s) red[t] += red[t + s]; __syncthreads(); }
  float var = red[0] / RDIM;
  float o = d * rsqrtf(var + 1e-5f) * g[t] + bta[t];
  uniq[(size_t)bm * RDIM + t] = (_Float16)o;
}

// ---------------- neighbor lists from sparse mask ----------------
__global__ void k_build_nbr(const unsigned char* __restrict__ mask,
                            int* __restrict__ nbr, int* __restrict__ cnt) {
  int m = blockIdx.x;
  if (threadIdx.x != 0) return;
  const unsigned char* row = mask + (size_t)m * MNODES;
  int c = 0;
  for (int n = 0; n < MNODES; ++n)
    if (row[n] && c < MAXNB) nbr[m * MAXNB + (c++)] = n;
  cnt[m] = c;
}

// ---------------- WMMA GEMM: C = A(rows x K) @ Wt^T + bias ----------------
// A: f16 row-major (rows x K); Wt: f16 (N x K) = W transposed.
// Block: 256 threads = 8 waves (2x4); block tile 64 rows x 128 cols.
// Wave tile 32x32 = 2x2 WMMA accumulators; LDS-double-buffered A tile, one
// barrier per 32-wide K step. EPI 0: store f32. EPI 1: gelu + store f16.
template<int EPI>
__global__ void k_gemm_wmma(const _Float16* __restrict__ A, const _Float16* __restrict__ Wt,
                            const float* __restrict__ bias, float* __restrict__ Cf,
                            _Float16* __restrict__ Ch, int K, int N) {
  const int wave = threadIdx.x >> 5;
  const int lane = threadIdx.x & 31;
  const int s  = lane >> 4;    // half-wave select
  const int rc = lane & 15;    // row (A) / col (B) within 16x16 tile
  const int wr = wave >> 2;    // wave row 0..1  -> 32-row slab
  const int wc = wave & 3;     // wave col 0..3  -> 32-col slab
  const int r0 = blockIdx.y * 64;
  const int c0 = blockIdx.x * 128 + wc * 32;

  __shared__ __align__(16) _Float16 As[2][64][40];  // double-buffered 64x32 A tile
  v8f acc[2][2] = {};

  const _Float16* wcol[2] = { Wt + (size_t)(c0 + rc) * K,
                              Wt + (size_t)(c0 + 16 + rc) * K };

  // cooperative staging: 2048 halves / 256 threads = one b128 each
  const int srow = threadIdx.x >> 2;          // 0..63
  const int soff = (threadIdx.x & 3) * 8;     // 0,8,16,24
  const _Float16* aRowPtr = A + (size_t)(r0 + srow) * K + soff;

  *(v8h*)&As[0][srow][soff] = *(const v8h*)aRowPtr;   // prologue tile
  int cur = 0;

  for (int k0 = 0; k0 < K; k0 += 32) {
    __syncthreads();  // publish As[cur]; fence prior reads of As[cur^1]
    const bool more = (k0 + 32 < K);
    v8h stage;
    if (more) stage = *(const v8h*)(aRowPtr + k0 + 32);

    // A fragments for both row tiles (reused across both col tiles)
    v16h afrag[2];
#pragma unroll
    for (int rt = 0; rt < 2; ++rt) {
      int lrow = wr * 32 + rt * 16 + rc;
      v8h alo = *(const v8h*)&As[cur][lrow][s * 8];
      v8h ahi = *(const v8h*)&As[cur][lrow][16 + s * 8];
#pragma unroll
      for (int i = 0; i < 8; ++i) { afrag[rt][i] = alo[i]; afrag[rt][8 + i] = ahi[i]; }
    }
#pragma unroll
    for (int ct = 0; ct < 2; ++ct) {
      v8h blo = *(const v8h*)(wcol[ct] + k0 + s * 8);
      v8h bhi = *(const v8h*)(wcol[ct] + k0 + 16 + s * 8);
      v16h bb;
#pragma unroll
      for (int i = 0; i < 8; ++i) { bb[i] = blo[i]; bb[8 + i] = bhi[i]; }
#pragma unroll
      for (int rt = 0; rt < 2; ++rt)
        acc[rt][ct] = __builtin_amdgcn_wmma_f32_16x16x32_f16(
            false, afrag[rt], false, bb, (short)0, acc[rt][ct], false, false);
    }
    if (more) *(v8h*)&As[cur ^ 1][srow][soff] = stage;
    cur ^= 1;
  }

#pragma unroll
  for (int ct = 0; ct < 2; ++ct) {
    int col = c0 + ct * 16 + rc;
    float bv = bias[col];
#pragma unroll
    for (int rt = 0; rt < 2; ++rt) {
#pragma unroll
      for (int r = 0; r < 8; ++r) {
        int row = r0 + wr * 32 + rt * 16 + s * 8 + r;  // C/D: VGPR r, lanes>=16 -> M=8+r
        float val = acc[rt][ct][r] + bv;
        if (EPI == 1) {
          const float kk = 0.7978845608028654f;
          float x3 = val * val * val;
          val = 0.5f * val * (1.0f + tanhf(kk * (val + 0.044715f * x3)));
          Ch[(size_t)row * N + col] = (_Float16)val;
        } else {
          Cf[(size_t)row * N + col] = val;
        }
      }
    }
  }
}

// ---------------- sparse tree attention (<=16 neighbors/node) ----------------
__global__ void k_attn(const float* __restrict__ qkv, const int* __restrict__ nbr,
                       const int* __restrict__ cnt, _Float16* __restrict__ ctx) {
  int bm = blockIdx.x, b = bm / MP, m = bm % MP, t = threadIdx.x;  // 128 threads
  if (m >= MNODES) { ctx[(size_t)bm * RDIM + t] = (_Float16)0.f; return; }
  __shared__ float qs[RDIM];
  __shared__ float ks[MAXNB][RDIM];
  __shared__ float vs[MAXNB][RDIM];
  const float* qrow = qkv + (size_t)bm * 384;
  qs[t] = qrow[t];
  int c = cnt[m]; if (c > MAXNB) c = MAXNB;
  for (int j = 0; j < c; ++j) {
    const float* nrow = qkv + (size_t)(b * MP + nbr[m * MAXNB + j]) * 384;
    ks[j][t] = nrow[128 + t];
    vs[j][t] = nrow[256 + t];
  }
  __syncthreads();
  int h = t >> 4, d = t & 15;
  float sc[MAXNB];
  float mx = -1e30f;
  for (int j = 0; j < c; ++j) {
    float sd = 0.f;
#pragma unroll
    for (int dd = 0; dd < HD; ++dd) sd += qs[h * HD + dd] * ks[j][h * HD + dd];
    sd *= 0.25f;                     // 1/sqrt(hd), hd=16
    sc[j] = sd; if (sd > mx) mx = sd;
  }
  float ssum = 0.f;
  for (int j = 0; j < c; ++j) { sc[j] = expf(sc[j] - mx); ssum += sc[j]; }
  float inv = (ssum > 0.f) ? 1.f / ssum : 0.f;
  float o = 0.f;
  for (int j = 0; j < c; ++j) o += sc[j] * vs[j][h * HD + d];
  ctx[(size_t)bm * RDIM + t] = (_Float16)(o * inv);
}

// ---------------- scatter attn_out, residual, LN2 -> f16 MLP input ----------------
__global__ void k_scatter_ln2(const float* __restrict__ x, const float* __restrict__ attn_out,
                              const int* __restrict__ idmap, const float* __restrict__ g,
                              const float* __restrict__ bta, float* __restrict__ xres,
                              _Float16* __restrict__ xln2) {
  int bi = blockIdx.x, t = threadIdx.x;
  int l = bi % L_LEVELS;
  int w = (bi / L_LEVELS) % HGRID;
  int h = (bi / (L_LEVELS * HGRID)) % HGRID;
  int b = bi / (L_LEVELS * HGRID * HGRID);
  int id = idmap[(h * HGRID + w) * L_LEVELS + l];
  size_t idx = ((((size_t)b * HGRID + h) * HGRID + w) * L_LEVELS + l) * RDIM + t;
  float v = x[idx] + attn_out[(size_t)(b * MP + id) * RDIM + t];
  xres[idx] = v;
  __shared__ float red[RDIM];
  red[t] = v; __syncthreads();
  for (int s = 64; s > 0; s >>= 1) { if (t < s) red[t] += red[t + s]; __syncthreads(); }
  float mean = red[0] / RDIM; __syncthreads();
  float dd = v - mean; red[t] = dd * dd; __syncthreads();
  for (int s = 64; s > 0; s >>= 1) { if (t < s) red[t] += red[t + s]; __syncthreads(); }
  float var = red[0] / RDIM;
  float o = dd * rsqrtf(var + 1e-5f) * g[t] + bta[t];
  xln2[((size_t)b * 1024 + (h * HGRID + w)) * NEMBD + l * RDIM + t] = (_Float16)o;
}

// ---------------- hierarchical s x s mean pooling per level ----------------
__global__ void k_pool(const float* __restrict__ mlp, float* __restrict__ pooled) {
  int bi = blockIdx.x, b = bi / MNODES, node = bi % MNODES, r = threadIdx.x;
  const int off[7] = {0, 1024, 1280, 1344, 1360, 1364, 1365};
  int lvl = 0;
  while (node >= off[lvl + 1]) ++lvl;
  int g = HGRID >> lvl, s = 1 << lvl;
  int loc = node - off[lvl], bh = loc / g, bw = loc % g;
  float sum = 0.f;
  for (int dh = 0; dh < s; ++dh)
    for (int dw = 0; dw < s; ++dw)
      sum += mlp[((size_t)b * 1024 + (bh * s + dh) * HGRID + (bw * s + dw)) * NEMBD
                 + lvl * RDIM + r];
  pooled[((size_t)b * MNODES + node) * RDIM + r] = sum / (float)(s * s);
}

// ---------------- final residual add (dx == block-constant pooled value) ----------------
__global__ void k_final(const float* __restrict__ xres, const float* __restrict__ pooled,
                        float* __restrict__ out) {
  int bi = blockIdx.x, t = threadIdx.x;
  int l = bi % L_LEVELS;
  int w = (bi / L_LEVELS) % HGRID;
  int h = (bi / (L_LEVELS * HGRID)) % HGRID;
  int b = bi / (L_LEVELS * HGRID * HGRID);
  const int off[6] = {0, 1024, 1280, 1344, 1360, 1364};
  int g = HGRID >> l;
  int node = off[l] + (h >> l) * g + (w >> l);
  size_t idx = ((((size_t)b * HGRID + h) * HGRID + w) * L_LEVELS + l) * RDIM + t;
  out[idx] = xres[idx] + pooled[((size_t)b * MNODES + node) * RDIM + t];
}

extern "C" void kernel_launch(void* const* d_in, const int* in_sizes, int n_in,
                              void* d_out, int out_size, void* d_ws, size_t ws_size,
                              hipStream_t stream) {
  (void)in_sizes; (void)n_in; (void)out_size; (void)ws_size;
  const float* x     = (const float*)d_in[0];
  const float* ln1_g = (const float*)d_in[1];
  const float* ln1_b = (const float*)d_in[2];
  const float* ln2_g = (const float*)d_in[3];
  const float* ln2_b = (const float*)d_in[4];
  const float* Wq    = (const float*)d_in[5];
  const float* bq    = (const float*)d_in[6];
  const float* Wk    = (const float*)d_in[7];
  const float* bk    = (const float*)d_in[8];
  const float* Wv    = (const float*)d_in[9];
  const float* bv    = (const float*)d_in[10];
  const float* Wo    = (const float*)d_in[11];
  const float* bo    = (const float*)d_in[12];
  const float* W1    = (const float*)d_in[13];
  const float* b1    = (const float*)d_in[14];
  const float* W2    = (const float*)d_in[15];
  const float* b2    = (const float*)d_in[16];
  const int*   idmap = (const int*)d_in[17];
  const int*   rep_h = (const int*)d_in[18];
  const int*   rep_w = (const int*)d_in[19];
  const int*   rep_l = (const int*)d_in[20];
  const unsigned char* mask = (const unsigned char*)d_in[21];

  char* ws = (char*)d_ws;
  size_t o = 0;
  auto alloc = [&](size_t bytes) -> char* {
    char* p = ws + o;
    o = (o + bytes + 255) & ~(size_t)255;
    return p;
  };
  _Float16* wqkvt = (_Float16*)alloc((size_t)384 * 128 * 2);
  _Float16* wot   = (_Float16*)alloc((size_t)128 * 128 * 2);
  _Float16* w1t   = (_Float16*)alloc((size_t)HID * NEMBD * 2);
  _Float16* w2t   = (_Float16*)alloc((size_t)NEMBD * HID * 2);
  float*    bqkv  = (float*)alloc(384 * 4);
  _Float16* uniq  = (_Float16*)alloc((size_t)ROWS_QKV * RDIM * 2);
  float*    qkv   = (float*)alloc((size_t)ROWS_QKV * 384 * 4);
  _Float16* ctx   = (_Float16*)alloc((size_t)ROWS_QKV * RDIM * 2);
  float*    aout  = (float*)alloc((size_t)ROWS_QKV * RDIM * 4);
  float*    xres  = (float*)alloc((size_t)BATCH * HGRID * HGRID * L_LEVELS * RDIM * 4);
  _Float16* xln2  = (_Float16*)alloc((size_t)ROWS_MLP * NEMBD * 2);
  _Float16* act   = (_Float16*)alloc((size_t)ROWS_MLP * HID * 2);
  float*    mlp   = (float*)alloc((size_t)ROWS_MLP * NEMBD * 4);
  float*    pooled= (float*)alloc((size_t)BATCH * MNODES * RDIM * 4);
  int*      nbr   = (int*)alloc((size_t)MNODES * MAXNB * 4);
  int*      cnt   = (int*)alloc((size_t)MNODES * 4);

  // weight prep
  int n128 = (128 * 128 + 255) / 256;
  k_transpose_f16<<<n128, 256, 0, stream>>>(Wq, wqkvt,             128, 128);
  k_transpose_f16<<<n128, 256, 0, stream>>>(Wk, wqkvt + 128 * 128, 128, 128);
  k_transpose_f16<<<n128, 256, 0, stream>>>(Wv, wqkvt + 256 * 128, 128, 128);
  k_transpose_f16<<<n128, 256, 0, stream>>>(Wo, wot,               128, 128);
  k_transpose_f16<<<(NEMBD * HID + 255) / 256, 256, 0, stream>>>(W1, w1t, NEMBD, HID);
  k_transpose_f16<<<(HID * NEMBD + 255) / 256, 256, 0, stream>>>(W2, w2t, HID, NEMBD);
  k_concat_bias<<<3, 128, 0, stream>>>(bq, bk, bv, bqkv);

  // attention path
  k_ln_gather<<<BATCH * MP, 128, 0, stream>>>(x, ln1_g, ln1_b, rep_h, rep_w, rep_l, uniq);
  k_build_nbr<<<MNODES, 32, 0, stream>>>(mask, nbr, cnt);
  dim3 gq(384 / 128, ROWS_QKV / 64);
  k_gemm_wmma<0><<<gq, 256, 0, stream>>>(uniq, wqkvt, bqkv, qkv, nullptr, 128, 384);
  k_attn<<<BATCH * MP, 128, 0, stream>>>(qkv, nbr, cnt, ctx);
  dim3 go(1, ROWS_QKV / 64);
  k_gemm_wmma<0><<<go, 256, 0, stream>>>(ctx, wot, bo, aout, nullptr, 128, 128);

  // residual + LN2 + MLP
  k_scatter_ln2<<<BATCH * HGRID * HGRID * L_LEVELS, 128, 0, stream>>>(
      x, aout, idmap, ln2_g, ln2_b, xres, xln2);
  dim3 g1(HID / 128, ROWS_MLP / 64);
  k_gemm_wmma<1><<<g1, 256, 0, stream>>>(xln2, w1t, b1, nullptr, act, NEMBD, HID);
  dim3 g2(NEMBD / 128, ROWS_MLP / 64);
  k_gemm_wmma<0><<<g2, 256, 0, stream>>>(act, w2t, b2, mlp, nullptr, HID, NEMBD);

  // pooling + final residual
  k_pool<<<BATCH * MNODES, 128, 0, stream>>>(mlp, pooled);
  k_final<<<BATCH * HGRID * HGRID * L_LEVELS, 128, 0, stream>>>(xres, pooled, (float*)d_out);
}
